// BGMABlock_42958262895027
// MI455X (gfx1250) — matrix operand order repaired and checked by reading, not compiled, not run on hardware
//
#include <hip/hip_runtime.h>
#include <cstdint>

// ---------------- constants ----------------
#define BSZ   8
#define CCH   512
#define HH    64
#define WW    64
#define HWSZ  4096            // 64*64
#define ESZ   16777216        // BSZ*CCH*HWSZ
#define LSEQ  512             // mamba sequence length == CCH
#define DMO   64              // d_model
#define DIN   128             // d_inner
#define NSTT  32              // d_state
#define NINST 4               // mamba instances: {avg-fwd, avg-bwd, max-fwd, max-bwd}

typedef __attribute__((ext_vector_type(16))) __bf16        v16bf;
typedef __attribute__((ext_vector_type(8)))  float         v8f;
typedef __attribute__((ext_vector_type(4)))  unsigned int  v4u;
typedef __attribute__((ext_vector_type(8)))  int           v8i;
typedef __attribute__((ext_vector_type(4)))  int           v4i;

__device__ __forceinline__ unsigned short f2bf(float f) {
  union { __bf16 h; unsigned short u; } c; c.h = (__bf16)f; return c.u;
}
__device__ __forceinline__ float sigmoidf_(float x) { return 1.0f / (1.0f + expf(-x)); }
__device__ __forceinline__ float siluf_(float x)    { return x * sigmoidf_(x); }
__device__ __forceinline__ float softplusf_(float x){ return (x > 20.0f) ? x : log1pf(expf(x)); }
__device__ __forceinline__ float geluf_(float x)    { return 0.5f * x * (1.0f + erff(x * 0.70710678118654752440f)); }

// ---------------- f32 -> bf16 convert ----------------
__global__ void k_cvt_bf16(const float* __restrict__ src, unsigned short* __restrict__ dst, int n) {
  int i = blockIdx.x * blockDim.x + threadIdx.x;
  if (i < n) dst[i] = f2bf(src[i]);
}

// ---------------- WMMA GEMM: D[o,m] = sum_k W[o,k]*X[k,m] (+epilogue) ----------------
// Wb: [512,512] bf16 row-major. Xb: [B][512][4096] bf16. grid: (32 mTiles, 4 oTiles, B)
// A-tile (weights) is fetched by the Tensor Data Mover directly into LDS:
//   2D D#: data_size=2B, tile 32(k,contig) x 128(rows), row stride 512 elems,
//   LDS dest linear row-major [128][32]  (identical layout to manual staging).
// mode 0: outf = gelu(acc + bias[o])
// mode 1: outbf = bf16((acc + bias[o]) * aux[idx])
// mode 2: outf = acc + bias[o] + aux[idx]
__global__ __launch_bounds__(256)
void k_gemm_wmma(const unsigned short* __restrict__ Wb,
                 const unsigned short* __restrict__ Xb,
                 const float* __restrict__ bias,
                 const float* __restrict__ aux,
                 float* __restrict__ outf,
                 unsigned short* __restrict__ outbf,
                 int mode)
{
  __shared__ unsigned short Al[128 * 32];   // [row o][k]   row-major (TDM destination)
  __shared__ unsigned short Bl[128 * 32];   // [col m][k]   (transposed stage)

  const int b    = blockIdx.z;
  const int o0   = blockIdx.y * 128;
  const int m0   = blockIdx.x * 128;
  const int tid  = threadIdx.x;
  const int lane = tid & 31;
  const int wid  = tid >> 5;
  const int wr   = wid >> 2;   // 0..1 : 64-row group
  const int wc   = wid & 3;    // 0..3 : 32-col group

  const unsigned short* X = Xb + (size_t)b * CCH * HWSZ;
  const unsigned lds_base = (unsigned)(size_t)(void*)Al;   // low 32 bits = LDS byte offset

  // D# group1 (constant across k-steps):
  //  [15:0] wg_mask=0 | [17:16] data_size=1 (2B) | pad/iter/barrier off
  //  tensor_dim0=512 @ [79:48], tensor_dim1=512 @ [111:80], tile_dim0=32 @ [127:112]
  //  tile_dim1=128 @ [143:128], tile_dim2=0, dim0_stride=512 @ [207:160], dim1_stride=0
  const v8i g1 = { 0x00010000, 0x02000000, 0x02000000, 0x00200000, 128, 512, 0, 0 };
  const v4i gz4 = { 0, 0, 0, 0 };
  const v8i gz8 = { 0, 0, 0, 0, 0, 0, 0, 0 };

  v8f acc[4][2];
#pragma unroll
  for (int i = 0; i < 4; ++i)
#pragma unroll
    for (int j = 0; j < 2; ++j) { v8f z = {0.f,0.f,0.f,0.f,0.f,0.f,0.f,0.f}; acc[i][j] = z; }

  for (int ks = 0; ks < CCH / 32; ++ks) {
    // ---- A tile via Tensor Data Mover (issued by wave 0 only; EXEC-independent DMA)
    if (wid == 0) {
      unsigned long long ga =
          (unsigned long long)(size_t)(Wb + (size_t)o0 * CCH + ks * 32);
      v4u g0 = { 1u,                                     // count=1 valid descriptor
                 lds_base,                               // lds_addr
                 (unsigned)ga,                           // global_addr[31:0]
                 ((unsigned)(ga >> 32) & 0x01FFFFFFu) | (2u << 30) };  // addr[56:32] | type=2
      __builtin_amdgcn_tensor_load_to_lds(g0, g1, gz4, gz4, gz8, 0);
    }
    // ---- B tile: global rows are k; scatter transposed to [m][k]
#pragma unroll
    for (int c = 0; c < 2; ++c) {
      int ch = tid * 2 + c;              // 0..511
      int k  = ch >> 4;                  // 0..31
      int m8 = (ch & 15) << 3;           // 0..120
      uint4 v = *(const uint4*)(X + (size_t)(ks * 32 + k) * HWSZ + m0 + m8);
      const unsigned short* pv = (const unsigned short*)&v;
#pragma unroll
      for (int j = 0; j < 8; ++j) Bl[(m8 + j) * 32 + k] = pv[j];
    }
    // prefetch next k-step's B rows into cache (global_prefetch_b8, no counter)
    if (ks + 1 < CCH / 32) {
      __builtin_prefetch((const void*)(X + (size_t)((ks + 1) * 32 + (tid >> 4)) * HWSZ +
                                       m0 + ((tid & 15) << 3)), 0, 1);
    }
    if (wid == 0) __builtin_amdgcn_s_wait_tensorcnt(0);  // TDM done before barrier signal
    __syncthreads();

    union FV { uint4 q[2]; v16bf v; };
    // B fragments: col = lane&15, K halves {kb..kb+7, kb+16..kb+23}, kb = 0/8
    FV bfr[2];
    const int mcol = lane & 15;
    const int kb   = (lane < 16) ? 0 : 8;
#pragma unroll
    for (int cb = 0; cb < 2; ++cb) {
      int m = wc * 32 + cb * 16 + mcol;
      bfr[cb].q[0] = *(const uint4*)(&Bl[m * 32 + kb]);
      bfr[cb].q[1] = *(const uint4*)(&Bl[m * 32 + kb + 16]);
    }
    // A fragments: row = base + lane&15, K halves {kh..kh+7, kh+16..kh+23}, kh = 0/8
    const int kh = (lane < 16) ? 0 : 8;
#pragma unroll
    for (int rb = 0; rb < 4; ++rb) {
      FV afr;
      int row = wr * 64 + rb * 16 + (lane & 15);
      afr.q[0] = *(const uint4*)(&Al[row * 32 + kh]);
      afr.q[1] = *(const uint4*)(&Al[row * 32 + 16 + kh]);
#pragma unroll
      for (int cb = 0; cb < 2; ++cb) {
        acc[rb][cb] = __builtin_amdgcn_wmma_f32_16x16x32_bf16(
            false, afr.v, false, bfr[cb].v, (short)0, acc[rb][cb], false, false);
      }
    }
    __syncthreads();
  }

  // epilogue: C/D layout: VGPR r -> M = (lane<16?0:8)+r, N = lane&15
  const int n     = lane & 15;
  const int rbase = (lane >> 4) << 3;
#pragma unroll
  for (int rb = 0; rb < 4; ++rb) {
#pragma unroll
    for (int cb = 0; cb < 2; ++cb) {
      int m = m0 + wc * 32 + cb * 16 + n;
#pragma unroll
      for (int r = 0; r < 8; ++r) {
        int o = o0 + wr * 64 + rb * 16 + rbase + r;
        size_t idx = ((size_t)b * CCH + o) * HWSZ + m;
        float v = acc[rb][cb][r] + bias[o];
        if (mode == 0) {
          outf[idx] = geluf_(v);
        } else if (mode == 1) {
          outbf[idx] = f2bf(v * aux[idx]);
        } else {
          outf[idx] = v + aux[idx];
        }
      }
    }
  }
}

// ---------------- per-(b,c) avg & max pool over HW ----------------
__global__ __launch_bounds__(256)
void k_pool(const float* __restrict__ x, float* __restrict__ xa, float* __restrict__ xm) {
  int bc = blockIdx.x;
  const float* p = x + (size_t)bc * HWSZ;
  float s = 0.f, mx = -3.4e38f;
  for (int i = threadIdx.x; i < HWSZ; i += 256) { float v = p[i]; s += v; mx = fmaxf(mx, v); }
  __shared__ float ss[256], sm[256];
  ss[threadIdx.x] = s; sm[threadIdx.x] = mx;
  __syncthreads();
  for (int st = 128; st > 0; st >>= 1) {
    if (threadIdx.x < st) {
      ss[threadIdx.x] += ss[threadIdx.x + st];
      sm[threadIdx.x] = fmaxf(sm[threadIdx.x], sm[threadIdx.x + st]);
    }
    __syncthreads();
  }
  if (threadIdx.x == 0) { xa[bc] = ss[0] * (1.0f / HWSZ); xm[bc] = sm[0]; }
}

// ---------------- build mamba input sequences s_in[inst][b][t][64] ----------------
__global__ void k_sin(const float* __restrict__ xa, const float* __restrict__ xm,
                      const float* __restrict__ avg_w, const float* __restrict__ avg_b,
                      const float* __restrict__ max_w, const float* __restrict__ max_b,
                      float* __restrict__ s_in) {
  int idx = blockIdx.x * blockDim.x + threadIdx.x;       // NINST*BSZ*LSEQ*64
  int j = idx & 63, rest = idx >> 6;
  int t = rest & 511; int b = (rest >> 9) & 7; int inst = rest >> 12;
  int ts = (inst & 1) ? (LSEQ - 1 - t) : t;              // flip for backward instances
  const float* pool = (inst < 2) ? xa : xm;
  const float* w    = (inst < 2) ? avg_w : max_w;
  const float* bb   = (inst < 2) ? avg_b : max_b;
  s_in[idx] = pool[b * LSEQ + ts] * w[j] + bb[j];
}

// ---------------- in_proj: xz[inst][b][t][256] ----------------
__global__ void k_inproj(const float* __restrict__ s_in,
                         const float* __restrict__ ipf, const float* __restrict__ ipb,
                         float* __restrict__ xz) {
  int idx = blockIdx.x * blockDim.x + threadIdx.x;       // NINST*BSZ*LSEQ*256
  int r = idx & 255; int sp = idx >> 8;                  // sp = ((inst*8+b)*512+t)
  int inst = sp >> 12;
  const float* ip = (inst & 1) ? ipb : ipf;              // [256,64]
  const float* s  = s_in + (size_t)sp * DMO;
  float a = 0.f;
#pragma unroll 8
  for (int j = 0; j < DMO; ++j) a += s[j] * ip[r * DMO + j];
  xz[idx] = a;
}

// ---------------- causal depthwise conv (K=4) + silu -> u ----------------
__global__ void k_convsilu(const float* __restrict__ xz,
                           const float* __restrict__ cwf, const float* __restrict__ cbf,
                           const float* __restrict__ cwb, const float* __restrict__ cbb,
                           float* __restrict__ u) {
  int idx = blockIdx.x * blockDim.x + threadIdx.x;       // NINST*BSZ*LSEQ*128
  int d = idx & 127; int sp = idx >> 7;
  int t = sp & 511; int inst = sp >> 12;
  const float* cw = (inst & 1) ? cwb : cwf;              // [128,4]
  const float* cb = (inst & 1) ? cbb : cbf;
  size_t base = (size_t)(sp - t);                        // ((inst*8+b)*512)
  float a = cb[d];
#pragma unroll
  for (int k = 0; k < 4; ++k) {
    int tt = t - 3 + k;
    if (tt >= 0) a += cw[d * 4 + k] * xz[(base + tt) * 256 + d];
  }
  u[idx] = siluf_(a);
}

// ---------------- x_proj: xdbl[inst][b][t][68] ----------------
__global__ void k_xproj(const float* __restrict__ u,
                        const float* __restrict__ xpf, const float* __restrict__ xpb,
                        float* __restrict__ xdbl) {
  int idx = blockIdx.x * blockDim.x + threadIdx.x;       // NINST*BSZ*LSEQ*68
  int r = idx % 68; int sp = idx / 68;
  int inst = sp >> 12;
  const float* xp = (inst & 1) ? xpb : xpf;              // [68,128]
  const float* up = u + (size_t)sp * DIN;
  float a = 0.f;
#pragma unroll 8
  for (int d = 0; d < DIN; ++d) a += up[d] * xp[r * DIN + d];
  xdbl[(size_t)sp * 68 + r] = a;
}

// ---------------- delta = softplus(dt @ dt_w.T + dt_b) ----------------
__global__ void k_delta(const float* __restrict__ xdbl,
                        const float* __restrict__ dwf, const float* __restrict__ dbf,
                        const float* __restrict__ dwb, const float* __restrict__ dbb,
                        float* __restrict__ delta) {
  int idx = blockIdx.x * blockDim.x + threadIdx.x;       // NINST*BSZ*LSEQ*128
  int d = idx & 127; int sp = idx >> 7;
  int inst = sp >> 12;
  const float* dw = (inst & 1) ? dwb : dwf;              // [128,4]
  const float* db = (inst & 1) ? dbb : dbf;
  float a = db[d];
#pragma unroll
  for (int r = 0; r < 4; ++r) a += xdbl[(size_t)sp * 68 + r] * dw[d * 4 + r];
  delta[idx] = softplusf_(a);
}

// ---------------- selective scan: one workgroup per (inst,b) sequence ----------------
// g[inst][b][t][d] = (ys + u*D) * silu(z)
__global__ __launch_bounds__(256)
void k_scan(const float* __restrict__ delta, const float* __restrict__ u,
            const float* __restrict__ xdbl, const float* __restrict__ xz,
            const float* __restrict__ Alogf, const float* __restrict__ Alogb,
            const float* __restrict__ Dpf, const float* __restrict__ Dpb,
            float* __restrict__ g)
{
  int inst = blockIdx.x >> 3, b = blockIdx.x & 7;
  const float* Alog = (inst & 1) ? Alogb : Alogf;        // [128,32]
  const float* Dp   = (inst & 1) ? Dpb   : Dpf;

  __shared__ float As[DIN * NSTT];
  __shared__ float Bs[NSTT], Cs[NSTT];
  __shared__ float part[256];
  for (int i = threadIdx.x; i < DIN * NSTT; i += 256) As[i] = -expf(Alog[i]);

  const int d  = threadIdx.x & 127;
  const int n0 = (threadIdx.x >> 7) * 16;                // 0 or 16
  float h[16];
#pragma unroll
  for (int i = 0; i < 16; ++i) h[i] = 0.f;
  const size_t sbase = ((size_t)inst * BSZ + b) * LSEQ;
  __syncthreads();

  for (int t = 0; t < LSEQ; ++t) {
    if (threadIdx.x < 64) {
      int nn = threadIdx.x & 31;
      float v = xdbl[(sbase + t) * 68 + ((threadIdx.x < 32) ? (4 + nn) : (36 + nn))];
      if (threadIdx.x < 32) Bs[nn] = v; else Cs[nn] = v;
    }
    __syncthreads();
    float dl  = delta[(sbase + t) * DIN + d];
    float uu  = u[(sbase + t) * DIN + d];
    float dbu = dl * uu;
    float acc = 0.f;
#pragma unroll
    for (int i = 0; i < 16; ++i) {
      int nn = n0 + i;
      float dA = __expf(dl * As[d * NSTT + nn]);
      h[i] = dA * h[i] + dbu * Bs[nn];
      acc += h[i] * Cs[nn];
    }
    part[threadIdx.x] = acc;
    __syncthreads();
    if (threadIdx.x < 128) {
      float y = part[threadIdx.x] + part[threadIdx.x + 128] + uu * Dp[d];
      float z = xz[(sbase + t) * 256 + 128 + d];
      g[(sbase + t) * DIN + d] = y * siluf_(z);
    }
  }
}

// ---------------- out_proj + bidir combine + gate + channel attention ----------------
__global__ __launch_bounds__(64)
void k_att(const float* __restrict__ g,
           const float* __restrict__ opf, const float* __restrict__ opb,
           const float* __restrict__ xa, const float* __restrict__ xm,
           const float* __restrict__ gate_w, const float* __restrict__ gate_b,
           const float* __restrict__ out_w, const float* __restrict__ out_b,
           float* __restrict__ att)
{
  int t = blockIdx.x & 511, b = blockIdx.x >> 9;
  int j = threadIdx.x;                                   // 0..63
  float oav = 0.f, omx = 0.f;
#pragma unroll
  for (int inst = 0; inst < NINST; ++inst) {
    const float* op = (inst & 1) ? opb : opf;            // [64,128]
    int ts = (inst & 1) ? (LSEQ - 1 - t) : t;            // flip back
    const float* gp = g + (((size_t)inst * BSZ + b) * LSEQ + ts) * DIN;
    float s = 0.f;
#pragma unroll 8
    for (int d = 0; d < DIN; ++d) s += gp[d] * op[j * DIN + d];
    if (inst < 2) oav += s; else omx += s;
  }
  float a  = sigmoidf_(gate_w[0] * xa[b * LSEQ + t] + gate_w[1] * xm[b * LSEQ + t] + gate_b[0]);
  float yy = a * oav + (1.0f - a) * omx;
  __shared__ float red[64];
  red[j] = yy * out_w[j];
  __syncthreads();
  for (int st = 32; st > 0; st >>= 1) {
    if (j < st) red[j] += red[j + st];
    __syncthreads();
  }
  if (j == 0) att[b * LSEQ + t] = sigmoidf_(red[0] + out_b[0]);
}

// ---------------- x *= att[b,c] ----------------
__global__ void k_scale(float* __restrict__ x, const float* __restrict__ att) {
  int idx = blockIdx.x * blockDim.x + threadIdx.x;
  int c = (idx >> 12) & 511, b = idx >> 21;
  x[idx] *= att[b * CCH + c];
}

// ---------------- generic depthwise conv ----------------
__global__ void k_dw(const float* __restrict__ src, float* __restrict__ dst,
                     const float* __restrict__ w, const float* __restrict__ bias,
                     int KH, int KW, int PH, int PW,
                     const float* __restrict__ addend, int accumulate) {
  int idx = blockIdx.x * blockDim.x + threadIdx.x;
  int xx0 = idx & 63, yy0 = (idx >> 6) & 63, c = (idx >> 12) & 511, b = idx >> 21;
  const float* sp = src + ((size_t)(b * CCH + c)) * HWSZ;
  float s = bias[c];
  for (int i = 0; i < KH; ++i) {
    int yy = yy0 + i - PH;
    if (yy < 0 || yy >= HH) continue;
    for (int jj = 0; jj < KW; ++jj) {
      int xx = xx0 + jj - PW;
      if (xx < 0 || xx >= WW) continue;
      s += w[(c * KH + i) * KW + jj] * sp[yy * WW + xx];
    }
  }
  if (addend) s += addend[idx];
  if (accumulate) dst[idx] += s; else dst[idx] = s;
}

// ---------------- host orchestration ----------------
extern "C" void kernel_launch(void* const* d_in, const int* in_sizes, int n_in,
                              void* d_out, int out_size, void* d_ws, size_t ws_size,
                              hipStream_t stream) {
  // flattened setup_inputs() order
  const float* inputs  = (const float*)d_in[0];
  const float* avg_w   = (const float*)d_in[1];
  const float* avg_b   = (const float*)d_in[2];
  const float* max_w   = (const float*)d_in[3];
  const float* max_b   = (const float*)d_in[4];
  const float* out_w   = (const float*)d_in[5];
  const float* out_b   = (const float*)d_in[6];
  const float* gate_w  = (const float*)d_in[7];
  const float* gate_b  = (const float*)d_in[8];
  // mf params at 9..17, mb at 18..26 (in_proj, conv_w, conv_b, x_proj, dt_w, dt_b, A_log, D, out_proj)
  const float* mf_ip = (const float*)d_in[9];
  const float* mf_cw = (const float*)d_in[10];
  const float* mf_cb = (const float*)d_in[11];
  const float* mf_xp = (const float*)d_in[12];
  const float* mf_dw = (const float*)d_in[13];
  const float* mf_db = (const float*)d_in[14];
  const float* mf_Al = (const float*)d_in[15];
  const float* mf_D  = (const float*)d_in[16];
  const float* mf_op = (const float*)d_in[17];
  const float* mb_ip = (const float*)d_in[18];
  const float* mb_cw = (const float*)d_in[19];
  const float* mb_cb = (const float*)d_in[20];
  const float* mb_xp = (const float*)d_in[21];
  const float* mb_dw = (const float*)d_in[22];
  const float* mb_db = (const float*)d_in[23];
  const float* mb_Al = (const float*)d_in[24];
  const float* mb_D  = (const float*)d_in[25];
  const float* mb_op = (const float*)d_in[26];
  const float* conv_w = (const float*)d_in[27];     // [512,512]
  const float* conv_b = (const float*)d_in[28];
  const float* w55   = (const float*)d_in[29]; const float* b55   = (const float*)d_in[30];
  const float* w1_7  = (const float*)d_in[31]; const float* b1_7  = (const float*)d_in[32];
  const float* w7_1  = (const float*)d_in[33]; const float* b7_1  = (const float*)d_in[34];
  const float* w1_11 = (const float*)d_in[35]; const float* b1_11 = (const float*)d_in[36];
  const float* w11_1 = (const float*)d_in[37]; const float* b11_1 = (const float*)d_in[38];
  const float* w1_21 = (const float*)d_in[39]; const float* b1_21 = (const float*)d_in[40];
  const float* w21_1 = (const float*)d_in[41]; const float* b21_1 = (const float*)d_in[42];

  float* out = (float*)d_out;

  // workspace carve-up
  char* ws = (char*)d_ws;
  size_t off = 0;
  auto carve = [&](size_t bytes) -> char* {
    char* p = ws + off; off += (bytes + 255) & ~(size_t)255; return p;
  };
  unsigned short* in_bf = (unsigned short*)carve((size_t)ESZ * 2);
  unsigned short* Wbf   = (unsigned short*)carve((size_t)CCH * CCH * 2);
  float* x_f    = (float*)carve((size_t)ESZ * 4);   // gelu(conv1x1(in)), later scaled in place
  float* x0_f   = (float*)carve((size_t)ESZ * 4);
  float* tmp_f  = (float*)carve((size_t)ESZ * 4);
  float* xs_f   = (float*)carve((size_t)ESZ * 4);
  unsigned short* xs_bf = (unsigned short*)carve((size_t)ESZ * 2);
  unsigned short* p_bf  = (unsigned short*)carve((size_t)ESZ * 2);
  float* xa_f   = (float*)carve((size_t)BSZ * CCH * 4);
  float* xm_f   = (float*)carve((size_t)BSZ * CCH * 4);
  float* sin_f  = (float*)carve((size_t)NINST * BSZ * LSEQ * DMO * 4);
  float* xz_f   = (float*)carve((size_t)NINST * BSZ * LSEQ * 256 * 4);
  float* u_f    = (float*)carve((size_t)NINST * BSZ * LSEQ * DIN * 4);
  float* xdbl_f = (float*)carve((size_t)NINST * BSZ * LSEQ * 68 * 4);
  float* del_f  = (float*)carve((size_t)NINST * BSZ * LSEQ * DIN * 4);
  float* g_f    = (float*)carve((size_t)NINST * BSZ * LSEQ * DIN * 4);
  float* att_f  = (float*)carve((size_t)BSZ * CCH * 4);

  const dim3 gGemm(HWSZ / 128, CCH / 128, BSZ);     // (32, 4, 8)

  // 1) bf16 copies of input tensor and 1x1 weight
  k_cvt_bf16<<<ESZ / 256, 256, 0, stream>>>(inputs, in_bf, ESZ);
  k_cvt_bf16<<<(CCH * CCH) / 256, 256, 0, stream>>>(conv_w, Wbf, CCH * CCH);

  // 2) x = gelu(conv1x1(inputs))  (WMMA GEMM, mode 0)
  k_gemm_wmma<<<gGemm, 256, 0, stream>>>(Wbf, in_bf, conv_b, nullptr, x_f, nullptr, 0);

  // 3) channel attention (mamba over channel sequences)
  k_pool<<<BSZ * CCH, 256, 0, stream>>>(x_f, xa_f, xm_f);
  k_sin<<<(NINST * BSZ * LSEQ * DMO) / 256, 256, 0, stream>>>(xa_f, xm_f, avg_w, avg_b, max_w, max_b, sin_f);
  k_inproj<<<(NINST * BSZ * LSEQ * 256) / 256, 256, 0, stream>>>(sin_f, mf_ip, mb_ip, xz_f);
  k_convsilu<<<(NINST * BSZ * LSEQ * DIN) / 256, 256, 0, stream>>>(xz_f, mf_cw, mf_cb, mb_cw, mb_cb, u_f);
  k_xproj<<<(NINST * BSZ * LSEQ * 68) / 256, 256, 0, stream>>>(u_f, mf_xp, mb_xp, xdbl_f);
  k_delta<<<(NINST * BSZ * LSEQ * DIN) / 256, 256, 0, stream>>>(xdbl_f, mf_dw, mf_db, mb_dw, mb_db, del_f);
  k_scan<<<NINST * BSZ, 256, 0, stream>>>(del_f, u_f, xdbl_f, xz_f, mf_Al, mb_Al, mf_D, mb_D, g_f);
  k_att<<<BSZ * LSEQ, 64, 0, stream>>>(g_f, mf_op, mb_op, xa_f, xm_f, gate_w, gate_b, out_w, out_b, att_f);
  k_scale<<<ESZ / 256, 256, 0, stream>>>(x_f, att_f);

  // 4) multi-scale depthwise convs; xs = x0 + x1 + x2 + x3
  k_dw<<<ESZ / 256, 256, 0, stream>>>(x_f,  x0_f, w55,   b55,   5, 5, 2, 2, nullptr, 0);
  k_dw<<<ESZ / 256, 256, 0, stream>>>(x0_f, tmp_f, w1_7,  b1_7,  1, 7, 0, 3, nullptr, 0);
  k_dw<<<ESZ / 256, 256, 0, stream>>>(tmp_f, xs_f, w7_1,  b7_1,  7, 1, 3, 0, x0_f,   0);  // xs = x1 + x0
  k_dw<<<ESZ / 256, 256, 0, stream>>>(x0_f, tmp_f, w1_11, b1_11, 1, 11, 0, 5, nullptr, 0);
  k_dw<<<ESZ / 256, 256, 0, stream>>>(tmp_f, xs_f, w11_1, b11_1, 11, 1, 5, 0, nullptr, 1); // xs += x2
  k_dw<<<ESZ / 256, 256, 0, stream>>>(x0_f, tmp_f, w1_21, b1_21, 1, 21, 0, 10, nullptr, 0);
  k_dw<<<ESZ / 256, 256, 0, stream>>>(tmp_f, xs_f, w21_1, b21_1, 21, 1, 10, 0, nullptr, 1); // xs += x3

  // 5) spatial_att = conv1x1(xs);  p = spatial_att * x  (fused, bf16 out, mode 1)
  k_cvt_bf16<<<ESZ / 256, 256, 0, stream>>>(xs_f, xs_bf, ESZ);
  k_gemm_wmma<<<gGemm, 256, 0, stream>>>(Wbf, xs_bf, conv_b, x_f, nullptr, p_bf, 1);

  // 6) out = conv1x1(p) + identity  (mode 2)
  k_gemm_wmma<<<gGemm, 256, 0, stream>>>(Wbf, p_bf, conv_b, inputs, out, nullptr, 2);

  (void)in_sizes; (void)n_in; (void)out_size; (void)ws_size;
}
